// Mamba2Enc_7447473291761
// MI455X (gfx1250) — matrix operand order repaired and checked by reading, not compiled, not run on hardware
//
#include <hip/hip_runtime.h>
#include <hip/hip_bf16.h>
#include <math.h>

// ---------------- model dims (from reference) ----------------
constexpr int B_ = 4, L_ = 2048, IN_ = 1024, D_ = 512;
constexpr int DI = 2 * D_;            // 1024
constexpr int DS = 128, HD = 64, K_ = 4;
constexpr int NH = DI / HD;           // 16
constexpr int CD = DI + 2 * DS;       // 1280
constexpr int DP = 2 * DI + 2 * DS + NH; // 2320
constexpr int NL = 2, NC = 2;
constexpr int M_ROWS = B_ * L_;       // 8192

typedef __attribute__((ext_vector_type(16))) _Float16 v16h;
typedef __attribute__((ext_vector_type(8)))  float    v8f;

// ---------------- CDNA5 async global->LDS copy (ASYNCcnt path) ----------------
__device__ __forceinline__ void async_copy_b128(void* lds_dst, const void* gsrc) {
    const unsigned int loff = (unsigned int)(uintptr_t)lds_dst;   // LDS byte offset (addr[31:0])
    const unsigned long long g = (unsigned long long)(uintptr_t)gsrc;
    asm volatile("global_load_async_to_lds_b128 %0, %1, off"
                 :: "v"(loff), "v"(g)
                 : "memory");
}
__device__ __forceinline__ void wait_asynccnt0() {
    asm volatile("s_wait_asynccnt 0x0" ::: "memory");
}

// ---------------- f32 -> f16 cast ----------------
__global__ void cast_f16_kernel(const float* __restrict__ s, _Float16* __restrict__ d, size_t n) {
    size_t i = (size_t)blockIdx.x * 256 + threadIdx.x;
    if (i < n) d[i] = (_Float16)s[i];
}

// ---------------- WMMA epilogue store of one 16x16 D tile ----------------
// MODE 0: plain f32 store ; MODE 1: +bias, exact GELU ; MODE 2: +resid (may alias C)
template <int MODE>
__device__ __forceinline__ void store_tile(float* __restrict__ C, const float* __restrict__ aux,
                                           const v8f& acc, int mbase, int col, int N) {
    if (col < N) {
        #pragma unroll
        for (int r = 0; r < 8; ++r) {
            const size_t idx = (size_t)(mbase + r) * N + col;
            float v = acc[r];
            if (MODE == 1) {
                v += aux[col];
                v = 0.5f * v * (1.0f + erff(v * 0.70710678118654752f));
            }
            if (MODE == 2) v += aux[idx];
            C[idx] = v;
        }
    }
}

// ---------------- generic WMMA GEMM: C(MxN) = A(MxK) @ W(NxK)^T ----------------
// Block tile 128(M) x 64(N); 8 waves, each owning a 32x32 D tile -> 4 WMMA per K-step.
template <int MODE>
__global__ __launch_bounds__(256)
void gemm_wmma_f16(const _Float16* __restrict__ A, const _Float16* __restrict__ W,
                   float* __restrict__ C, const float* __restrict__ aux,
                   int M, int N, int Kd) {
    __shared__ __align__(16) _Float16 As[128 * 32];   // 8 KB
    __shared__ __align__(16) _Float16 Bs[64 * 32];    // 4 KB

    const int tid  = threadIdx.x;
    const int lane = tid & 31;
    const int wave = tid >> 5;
    const int wm   = wave & 3;    // 4 in M, 32 rows each
    const int wn   = wave >> 2;   // 2 in N, 32 cols each
    const int m0   = blockIdx.y * 128;
    const int n0   = blockIdx.x * 64;
    const int hi   = lane >> 4;   // K-half selector per documented 16-bit layouts
    const int l15  = lane & 15;

    v8f acc00 = {}, acc01 = {}, acc10 = {}, acc11 = {};

    for (int k0 = 0; k0 < Kd; k0 += 32) {
        // --- stage A tile (128x32 halfs) via async global->LDS, 2x16B per thread ---
        #pragma unroll
        for (int c = 0; c < 2; ++c) {
            const int id  = tid + c * 256;
            const int row = id >> 2, kc = (id & 3) * 8;
            async_copy_b128(&As[row * 32 + kc], &A[(size_t)(m0 + row) * Kd + k0 + kc]);
        }
        // --- stage B tile (64x32 halfs), zero-fill rows past N ---
        {
            const int row = tid >> 2, kc = (tid & 3) * 8;
            if (n0 + row < N) {
                async_copy_b128(&Bs[row * 32 + kc], &W[(size_t)(n0 + row) * Kd + k0 + kc]);
            } else {
                float4 z = {};
                *reinterpret_cast<float4*>(&Bs[row * 32 + kc]) = z;
            }
        }
        wait_asynccnt0();
        __syncthreads();

        // A frag: lanes 0-15 hold K 0-7,16-23 ; lanes 16-31 hold K 8-15,24-31 (row M = lane&15)
        union { v16h v; float4 q[2]; } a0, a1, b0, b1;
        const float4* Ar0 = reinterpret_cast<const float4*>(&As[(wm * 32 + l15) * 32]);
        const float4* Ar1 = reinterpret_cast<const float4*>(&As[(wm * 32 + 16 + l15) * 32]);
        a0.q[0] = Ar0[hi];     a0.q[1] = Ar0[hi + 2];
        a1.q[0] = Ar1[hi];     a1.q[1] = Ar1[hi + 2];
        // B frag: lanes 0-15 hold K 0-7,16-23 ; lanes 16-31 hold K 8-15,24-31 (col N = lane&15)
        const float4* Br0 = reinterpret_cast<const float4*>(&Bs[(wn * 32 + l15) * 32]);
        const float4* Br1 = reinterpret_cast<const float4*>(&Bs[(wn * 32 + 16 + l15) * 32]);
        b0.q[0] = Br0[hi];     b0.q[1] = Br0[hi + 2];
        b1.q[0] = Br1[hi];     b1.q[1] = Br1[hi + 2];

        acc00 = __builtin_amdgcn_wmma_f32_16x16x32_f16(false, a0.v, false, b0.v, (short)0, acc00, false, false);
        acc01 = __builtin_amdgcn_wmma_f32_16x16x32_f16(false, a0.v, false, b1.v, (short)0, acc01, false, false);
        acc10 = __builtin_amdgcn_wmma_f32_16x16x32_f16(false, a1.v, false, b0.v, (short)0, acc10, false, false);
        acc11 = __builtin_amdgcn_wmma_f32_16x16x32_f16(false, a1.v, false, b1.v, (short)0, acc11, false, false);
        __syncthreads();
    }

    // D layout: VGPR r -> M = mbase + r, mbase includes hi*8 ; N = lane&15
    const int mb = m0 + wm * 32 + hi * 8;
    const int cb = n0 + wn * 32 + l15;
    store_tile<MODE>(C, aux, acc00, mb,      cb,      N);
    store_tile<MODE>(C, aux, acc01, mb,      cb + 16, N);
    store_tile<MODE>(C, aux, acc10, mb + 16, cb,      N);
    store_tile<MODE>(C, aux, acc11, mb + 16, cb + 16, N);
}

// ---------------- LayerNorm: one block per row of 512, emits f16 ----------------
__global__ __launch_bounds__(256)
void ln_kernel(const float* __restrict__ h, const float* __restrict__ w,
               const float* __restrict__ b, _Float16* __restrict__ o) {
    const int r = blockIdx.x, tid = threadIdx.x;
    __shared__ float rs[256], rq[256];
    const float a0 = h[(size_t)r * D_ + tid];
    const float a1 = h[(size_t)r * D_ + tid + 256];
    rs[tid] = a0 + a1;
    rq[tid] = a0 * a0 + a1 * a1;
    __syncthreads();
    for (int s = 128; s > 0; s >>= 1) {
        if (tid < s) { rs[tid] += rs[tid + s]; rq[tid] += rq[tid + s]; }
        __syncthreads();
    }
    const float m   = rs[0] * (1.0f / D_);
    const float var = rq[0] * (1.0f / D_) - m * m;
    const float inv = rsqrtf(var + 1e-5f);
    o[(size_t)r * D_ + tid]       = (_Float16)((a0 - m) * inv * w[tid] + b[tid]);
    o[(size_t)r * D_ + tid + 256] = (_Float16)((a1 - m) * inv * w[tid + 256] + b[tid + 256]);
}

// ---------------- causal conv (K=4) + SiLU over xBC channels ----------------
__global__ void conv_silu_kernel(const float* __restrict__ zx, const float* __restrict__ cw,
                                 const float* __restrict__ cb, float* __restrict__ out) {
    const size_t idx = (size_t)blockIdx.x * 256 + threadIdx.x;
    if (idx >= (size_t)M_ROWS * CD) return;
    const int c = (int)(idx % CD);
    const int r = (int)(idx / CD);
    const int t = r % L_;
    float y = cb[c];
    #pragma unroll
    for (int k = 0; k < K_; ++k) {
        const int tt = t - (K_ - 1) + k;
        if (tt >= 0)
            y += zx[(size_t)(r - (K_ - 1) + k) * DP + DI + c] * cw[c * K_ + k];
    }
    y = y / (1.0f + expf(-y)); // SiLU
    out[(size_t)r * CD + c] = y;
}

// ---------------- dt = softplus(raw + bias); dA = exp(-exp(Alog)*dt) ----------------
__global__ void dt_kernel(const float* __restrict__ zx, const float* __restrict__ dtb,
                          const float* __restrict__ Alog,
                          float* __restrict__ dtv, float* __restrict__ dAv) {
    const int idx = blockIdx.x * 256 + threadIdx.x;
    if (idx >= M_ROWS * NH) return;
    const int hh = idx & (NH - 1);
    const int r  = idx >> 4;
    float raw = zx[(size_t)r * DP + DI + CD + hh] + dtb[hh];
    float dt  = (raw > 20.0f) ? raw : log1pf(expf(raw));
    dtv[idx] = dt;
    dAv[idx] = expf(-expf(Alog[hh]) * dt);
}

// ---------------- selective scan: one block per (b, head); 2048 sequential steps ----------------
// state h[64][128] f32 -> 32 regs per thread (256 threads); thread t: p = t>>2, n in [(t&3)*32, +32)
__global__ __launch_bounds__(256)
void scan_kernel(const float* __restrict__ conv, const float* __restrict__ dtv,
                 const float* __restrict__ dAv, const float* __restrict__ Dp,
                 float* __restrict__ y) {
    const int blk = blockIdx.x;
    const int b   = blk / NH;
    const int hh  = blk % NH;
    const int tid = threadIdx.x;
    const int p   = tid >> 2;
    const int n0  = (tid & 3) * 32;

    __shared__ __align__(16) float xs[HD], Bs[DS], Cs[DS];
    __shared__ float sdA, sdt;

    float s[32];
    #pragma unroll
    for (int j = 0; j < 32; ++j) s[j] = 0.0f;

    const float Dv = Dp[hh];

    for (int t = 0; t < L_; ++t) {
        const size_t base = (size_t)(b * L_ + t) * CD;
        // stage x(64) + B(128) + C(128) = 80 x 16B chunks via async global->LDS
        if (tid < 80) {
            if (tid < 16) {
                async_copy_b128(&xs[tid * 4], &conv[base + hh * HD + tid * 4]);
            } else if (tid < 48) {
                const int c = tid - 16;
                async_copy_b128(&Bs[c * 4], &conv[base + DI + c * 4]);
            } else {
                const int c = tid - 48;
                async_copy_b128(&Cs[c * 4], &conv[base + DI + DS + c * 4]);
            }
        }
        if (tid == 0) {
            const size_t di = (size_t)(b * L_ + t) * NH + hh;
            sdA = dAv[di];
            sdt = dtv[di];
        }
        wait_asynccnt0();
        __syncthreads();

        const float dA   = sdA;
        const float xp   = xs[p];
        const float dtxp = sdt * xp;
        float yacc = 0.0f;
        #pragma unroll
        for (int j = 0; j < 32; ++j) {
            const int n = n0 + j;
            s[j] = s[j] * dA + dtxp * Bs[n];
            yacc += s[j] * Cs[n];
        }
        // reduce across the 4 lanes sharing row p (consecutive lanes, wave32)
        yacc += __shfl_down(yacc, 2, 4);
        yacc += __shfl_down(yacc, 1, 4);
        if ((tid & 3) == 0)
            y[(size_t)(b * L_ + t) * DI + hh * HD + p] = yacc + Dv * xp;
        __syncthreads();
    }
}

// ---------------- gate: g = y*silu(z); RMSNorm * gw ; emit f16 ----------------
__global__ __launch_bounds__(256)
void gate_rmsnorm_kernel(const float* __restrict__ zx, const float* __restrict__ y,
                         const float* __restrict__ gw, _Float16* __restrict__ gh) {
    const int r = blockIdx.x, tid = threadIdx.x;
    __shared__ float red[256];
    float g[4];
    float ss = 0.0f;
    #pragma unroll
    for (int j = 0; j < 4; ++j) {
        const int i  = tid + j * 256;
        const float z  = zx[(size_t)r * DP + i];
        const float yv = y[(size_t)r * DI + i];
        const float gv = yv * (z / (1.0f + expf(-z)));
        g[j] = gv;
        ss  += gv * gv;
    }
    red[tid] = ss;
    __syncthreads();
    for (int s = 128; s > 0; s >>= 1) {
        if (tid < s) red[tid] += red[tid + s];
        __syncthreads();
    }
    const float scale = rsqrtf(red[0] * (1.0f / DI) + 1e-5f);
    #pragma unroll
    for (int j = 0; j < 4; ++j) {
        const int i = tid + j * 256;
        gh[(size_t)r * DI + i] = (_Float16)(g[j] * scale * gw[i]);
    }
}

// ---------------- classifier: out[r, c] = h[r,:] . cls_w[c,:] + cls_b[c] ----------------
__global__ void cls_kernel(const float* __restrict__ h, const float* __restrict__ cw,
                           const float* __restrict__ cb, float* __restrict__ out) {
    const int r = blockIdx.x * 256 + threadIdx.x;
    if (r >= M_ROWS) return;
    float a0 = cb[0], a1 = cb[1];
    for (int k = 0; k < D_; ++k) {
        const float hv = h[(size_t)r * D_ + k];
        a0 += hv * cw[k];
        a1 += hv * cw[D_ + k];
    }
    out[(size_t)r * NC + 0] = a0;
    out[(size_t)r * NC + 1] = a1;
}

// ---------------- host launch ----------------
extern "C" void kernel_launch(void* const* d_in, const int* in_sizes, int n_in,
                              void* d_out, int out_size, void* d_ws, size_t ws_size,
                              hipStream_t stream) {
    const float* d_x       = (const float*)d_in[0];
    const float* d_fc1_w   = (const float*)d_in[1];
    const float* d_fc1_b   = (const float*)d_in[2];
    const float* d_ln_w    = (const float*)d_in[3];
    const float* d_ln_b    = (const float*)d_in[4];
    const float* d_inp_w   = (const float*)d_in[5];
    const float* d_conv_w  = (const float*)d_in[6];
    const float* d_conv_b  = (const float*)d_in[7];
    const float* d_dt_b    = (const float*)d_in[8];
    const float* d_Alog    = (const float*)d_in[9];
    const float* d_Dp      = (const float*)d_in[10];
    const float* d_gnw     = (const float*)d_in[11];
    const float* d_outp_w  = (const float*)d_in[12];
    const float* d_cls_w   = (const float*)d_in[13];
    const float* d_cls_b   = (const float*)d_in[14];
    float* d_o = (float*)d_out;

    char* ws = (char*)d_ws;
    size_t off = 0;
    auto take = [&](size_t bytes) -> void* {
        void* p = ws + off;
        off = (off + bytes + 255) & ~(size_t)255;
        return p;
    };

    _Float16* xh  = (_Float16*)take((size_t)M_ROWS * IN_ * 2);
    _Float16* w1h = (_Float16*)take((size_t)D_ * IN_ * 2);
    _Float16* wih = (_Float16*)take((size_t)NL * DP * D_ * 2);
    _Float16* woh = (_Float16*)take((size_t)NL * D_ * DI * 2);
    float*    hb  = (float*)   take((size_t)M_ROWS * D_ * 4);
    _Float16* lnh = (_Float16*)take((size_t)M_ROWS * D_ * 2);
    float*    zx  = (float*)   take((size_t)M_ROWS * DP * 4);
    float*    cvb = (float*)   take((size_t)M_ROWS * CD * 4);
    float*    dtv = (float*)   take((size_t)M_ROWS * NH * 4);
    float*    dAv = (float*)   take((size_t)M_ROWS * NH * 4);
    float*    yb  = (float*)   take((size_t)M_ROWS * DI * 4);
    _Float16* gh  = (_Float16*)take((size_t)M_ROWS * DI * 2);

    auto cast = [&](const float* s, _Float16* d, size_t n) {
        cast_f16_kernel<<<(unsigned)((n + 255) / 256), 256, 0, stream>>>(s, d, n);
    };
    cast(d_x,      xh,  (size_t)M_ROWS * IN_);
    cast(d_fc1_w,  w1h, (size_t)D_ * IN_);
    cast(d_inp_w,  wih, (size_t)NL * DP * D_);
    cast(d_outp_w, woh, (size_t)NL * D_ * DI);

    // fc1 + GELU: (8192 x 1024) @ (512 x 1024)^T -> hb
    {
        dim3 g((D_ + 63) / 64, M_ROWS / 128);
        gemm_wmma_f16<1><<<g, 256, 0, stream>>>(xh, w1h, hb, d_fc1_b, M_ROWS, D_, IN_);
    }

    for (int l = 0; l < NL; ++l) {
        ln_kernel<<<M_ROWS, 256, 0, stream>>>(hb, d_ln_w + l * D_, d_ln_b + l * D_, lnh);

        // in_proj: (8192 x 512) @ (2320 x 512)^T -> zx
        {
            dim3 g((DP + 63) / 64, M_ROWS / 128);
            gemm_wmma_f16<0><<<g, 256, 0, stream>>>(lnh, wih + (size_t)l * DP * D_, zx,
                                                    nullptr, M_ROWS, DP, D_);
        }

        {
            const size_t total = (size_t)M_ROWS * CD;
            conv_silu_kernel<<<(unsigned)((total + 255) / 256), 256, 0, stream>>>(
                zx, d_conv_w + (size_t)l * CD * K_, d_conv_b + l * CD, cvb);
        }

        dt_kernel<<<(M_ROWS * NH + 255) / 256, 256, 0, stream>>>(
            zx, d_dt_b + l * NH, d_Alog + l * NH, dtv, dAv);

        scan_kernel<<<B_ * NH, 256, 0, stream>>>(cvb, dtv, dAv, d_Dp + l * NH, yb);

        gate_rmsnorm_kernel<<<M_ROWS, 256, 0, stream>>>(zx, yb, d_gnw + l * DI, gh);

        // out_proj + residual: (8192 x 1024) @ (512 x 1024)^T + hb -> hb
        {
            dim3 g((D_ + 63) / 64, M_ROWS / 128);
            gemm_wmma_f16<2><<<g, 256, 0, stream>>>(gh, woh + (size_t)l * D_ * DI, hb,
                                                    hb, M_ROWS, D_, DI);
        }
    }

    cls_kernel<<<(M_ROWS + 255) / 256, 256, 0, stream>>>(hb, d_cls_w, d_cls_b, d_o);
}
